// differentiableTokenizerVisionTransformer_56298431316560
// MI455X (gfx1250) — matrix operand order repaired and checked by reading, not compiled, not run on hardware
//
#include <hip/hip_runtime.h>
#include <cstdint>
#include <cstddef>

// ---------------------------------------------------------------------------
// Types / helpers
// ---------------------------------------------------------------------------
typedef __attribute__((ext_vector_type(16))) __bf16 v16bf;
typedef __attribute__((ext_vector_type(8)))  float  v8f;
typedef __attribute__((ext_vector_type(4)))  int    v4i;
typedef unsigned short u16;
typedef unsigned int   u32;

struct U32B { uint4 a, b; };

static __device__ __forceinline__ v16bf frag2(const u16* p0, const u16* p1) {
  U32B u;
  u.a = *(const uint4*)p0;
  u.b = *(const uint4*)p1;
  return __builtin_bit_cast(v16bf, u);
}

static __device__ __forceinline__ u16 f2bf(float f) {
  u32 u = __builtin_bit_cast(u32, f);
  u32 r = u + 0x7FFFu + ((u >> 16) & 1u);   // round-to-nearest-even
  return (u16)(r >> 16);
}

// --- CDNA5 async global->LDS path (guarded; sync fallback) -----------------
#if defined(__HIP_DEVICE_COMPILE__) && \
    __has_builtin(__builtin_amdgcn_global_load_async_to_lds_b128) && \
    __has_builtin(__builtin_amdgcn_s_wait_asynccnt)
#define ASYNC_LDS 1
#else
#define ASYNC_LDS 0
#endif

#if ASYNC_LDS
typedef __attribute__((address_space(1))) v4i* as1p;   // global (prints as __device__)
typedef __attribute__((address_space(3))) v4i* as3p;   // LDS
#endif

static __device__ __forceinline__ void tile_cp16(const u16* g, u16* l) {
#if ASYNC_LDS
  __builtin_amdgcn_global_load_async_to_lds_b128((as1p)(v4i*)(u16*)g,
                                                 (as3p)(v4i*)l, 0, 0);
#else
  *(uint4*)l = *(const uint4*)g;
#endif
}

template<int N>
static __device__ __forceinline__ void tile_wait() {
#if ASYNC_LDS
  __builtin_amdgcn_s_wait_asynccnt(N);
#endif
}

// Problem constants
#define BB_   32
#define SS_   256
#define DD_   768
#define LL_   12
#define NHH_  12
#define HD_   64
#define NN_   257          // tokens per image
#define MM_   (BB_*NN_)    // 8224 rows
#define NPAD_ 272          // padded token count (17 tiles of 16)
#define KPAD_ 288          // padded K for P@V (multiple of 32)
#define MLP_  3072
#define HW_   50176
#define NC_   1000

// ---------------------------------------------------------------------------
// Small utility kernels
// ---------------------------------------------------------------------------
__global__ void zero_f32(float* p, int n) {
  int i = blockIdx.x * 256 + threadIdx.x;
  if (i < n) p[i] = 0.f;
}

__global__ void seg_sum_kernel(const float* __restrict__ img,
                               const int* __restrict__ seg,
                               float* __restrict__ segsum) {
  long long idx = (long long)blockIdx.x * 256 + threadIdx.x;
  if (idx >= (long long)BB_ * HW_) return;
  int hw = (int)(idx % HW_);
  int b  = (int)(idx / HW_);
  int sid = seg[idx];
  float* dst = segsum + ((long long)b * SS_ + sid) * 3;
  const float* src = img + (long long)b * 3 * HW_ + hw;
  atomicAdd(dst + 0, src[0]);
  atomicAdd(dst + 1, src[HW_]);
  atomicAdd(dst + 2, src[2 * HW_]);
}

__global__ void tokens_kernel(const float* __restrict__ segsum,
                              const float* __restrict__ w_fe,
                              const float* __restrict__ b_fe,
                              const float* __restrict__ cls_tok,
                              const float* __restrict__ pos,
                              float* __restrict__ x) {
  long long idx = (long long)blockIdx.x * 256 + threadIdx.x;
  if (idx >= (long long)MM_ * DD_) return;
  int d = (int)(idx % DD_);
  long long row = idx / DD_;
  int t = (int)(row % NN_);
  int b = (int)(row / NN_);
  float v;
  if (t == 0) {
    v = cls_tok[d] + pos[d];
  } else {
    int s = t - 1;
    const float* ss = segsum + ((long long)b * SS_ + s) * 3;
    const float inv = 1.f / (float)HW_;
    v = ss[0] * inv * w_fe[d] + ss[1] * inv * w_fe[DD_ + d] +
        ss[2] * inv * w_fe[2 * DD_ + d] + b_fe[d] + pos[(long long)t * DD_ + d];
  }
  x[idx] = v;
}

// fp32 [L][K][N] -> bf16 transposed [L][N][K]
__global__ void cvt_transpose(const float* __restrict__ src,
                              u16* __restrict__ dst,
                              int K, int N, long long total) {
  long long idx = (long long)blockIdx.x * 256 + threadIdx.x;
  if (idx >= total) return;
  long long per = (long long)K * N;
  long long l = idx / per;
  long long r = idx % per;
  int k = (int)(r % K);
  int n = (int)(r / K);
  dst[idx] = f2bf(src[l * per + (long long)k * N + n]);
}

// qkv(bf16) V-part -> vT[bh][f][key] bf16, key padded to KPAD_ with zeros
__global__ void build_vt(const u16* __restrict__ qkv, u16* __restrict__ vT) {
  long long idx = (long long)blockIdx.x * 256 + threadIdx.x;
  const long long total = (long long)BB_ * NHH_ * HD_ * KPAD_;
  if (idx >= total) return;
  int key = (int)(idx % KPAD_);
  int f   = (int)((idx / KPAD_) % HD_);
  int bh  = (int)(idx / ((long long)KPAD_ * HD_));
  int h = bh % NHH_, b = bh / NHH_;
  u16 val = 0;
  if (key < NN_) val = qkv[((long long)(b * NN_ + key)) * (3 * DD_) + 2 * DD_ + h * HD_ + f];
  vT[idx] = val;
}

// ---------------------------------------------------------------------------
// LayerNorm over 768 columns, one wave per row
// ---------------------------------------------------------------------------
template<bool OUTF32>
__global__ __launch_bounds__(256) void ln_rows(const float* __restrict__ x,
                                               long long rowStride, int rows,
                                               const float* __restrict__ g,
                                               const float* __restrict__ bb,
                                               void* __restrict__ out,
                                               long long outStride) {
  int row = blockIdx.x * 8 + (threadIdx.x >> 5);
  if (row >= rows) return;
  int lane = threadIdx.x & 31;
  const float* xr = x + (long long)row * rowStride;
  float v[24];
  float s = 0.f;
#pragma unroll
  for (int i = 0; i < 24; ++i) { v[i] = xr[lane + i * 32]; s += v[i]; }
#pragma unroll
  for (int m = 16; m; m >>= 1) s += __shfl_xor(s, m, 32);
  float mean = s * (1.f / 768.f);
  float s2 = 0.f;
#pragma unroll
  for (int i = 0; i < 24; ++i) { float d = v[i] - mean; s2 += d * d; }
#pragma unroll
  for (int m = 16; m; m >>= 1) s2 += __shfl_xor(s2, m, 32);
  float inv = rsqrtf(s2 * (1.f / 768.f) + 1e-6f);
#pragma unroll
  for (int i = 0; i < 24; ++i) {
    int d = lane + i * 32;
    float y = (v[i] - mean) * inv * g[d] + bb[d];
    if (OUTF32) ((float*)out)[(long long)row * outStride + d] = y;
    else        ((u16*)out)[(long long)row * outStride + d] = f2bf(y);
  }
}

// ---------------------------------------------------------------------------
// Generic WMMA bf16 GEMM:  C[M,N] = A[M,K](bf16,row) * Bt[N,K](bf16) + epilogue
// Block: 256 threads (8 waves), tile 128(M) x 64(N); each wave 16x64.
// Double-buffered LDS with async global->LDS prefetch (CDNA5 ASYNCcnt path).
// EP: 1 = f32 out + bias + residual ; 2 = bf16 out + bias ; 3 = bf16 gelu(out+bias)
// ---------------------------------------------------------------------------
template<int EP>
__global__ __launch_bounds__(256) void gemm_bf16(
    const u16* __restrict__ A, long long lda, long long strideA,
    const u16* __restrict__ Bt, long long ldb, long long strideB,
    const float* __restrict__ bias,
    void* __restrict__ Cout, long long ldc,
    const float* __restrict__ Res,
    long long sC_outer, long long sC_inner, int c_inner,
    int M, int Mstore, int N, int K) {
  __shared__ __align__(16) u16 sA[2][128][40];
  __shared__ __align__(16) u16 sB[2][64][40];
  int bz = blockIdx.z;
  const u16* Ab = A + (long long)bz * strideA;
  const u16* Bb = Bt + (long long)bz * strideB;
  long long coff = (long long)(bz / c_inner) * sC_outer + (long long)(bz % c_inner) * sC_inner;
  int m0 = blockIdx.y * 128;
  int n0 = blockIdx.x * 64;
  int tid = threadIdx.x;
  int wave = tid >> 5, lane = tid & 31;
  int l16 = lane & 15, lh = lane >> 4;

  v8f acc[4];
#pragma unroll
  for (int t = 0; t < 4; ++t) acc[t] = v8f{0, 0, 0, 0, 0, 0, 0, 0};

  // issue one 128x32 A tile + 64x32 B tile into LDS buffer `buf`
  // (3 x 16-byte async copies per thread)
  auto issue = [&](int buf, int k0) {
#pragma unroll
    for (int c = 0; c < 2; ++c) {
      int ch = tid + c * 256;
      int r = ch >> 2, c8 = ch & 3;
      int gm = m0 + r; if (gm > M - 1) gm = M - 1;
      tile_cp16(&Ab[(long long)gm * lda + k0 + c8 * 8], &sA[buf][r][c8 * 8]);
    }
    int r = tid >> 2, c8 = tid & 3;
    int gn = n0 + r; if (gn > N - 1) gn = N - 1;
    tile_cp16(&Bb[(long long)gn * ldb + k0 + c8 * 8], &sB[buf][r][c8 * 8]);
  };

  const int ktiles = K >> 5;
  issue(0, 0);
  for (int t = 0; t < ktiles; ++t) {
    int buf = t & 1;
    if (t + 1 < ktiles) {
      issue(buf ^ 1, (t + 1) << 5);   // prefetch next tile into other buffer
      tile_wait<3>();                 // wait only for the PREVIOUS tile's copies
    } else {
      tile_wait<0>();
    }
    __syncthreads();

    const u16* arow = &sA[buf][wave * 16 + l16][0];
    v16bf af = frag2(arow + lh * 8, arow + 16 + lh * 8);
#pragma unroll
    for (int tt = 0; tt < 4; ++tt) {
      const u16* brow = &sB[buf][tt * 16 + l16][lh * 8];
      v16bf bf = frag2(brow, brow + 16);
      acc[tt] = __builtin_amdgcn_wmma_f32_16x16x32_bf16(false, af, false, bf,
                                                        (short)0, acc[tt], false, false);
    }
    __syncthreads();   // buffer we just read gets overwritten 2 tiles from now
  }

  int mbase = m0 + wave * 16 + lh * 8;
#pragma unroll
  for (int t = 0; t < 4; ++t) {
    int n = n0 + t * 16 + l16;
    float bv = bias ? bias[n] : 0.0f;
#pragma unroll
    for (int r = 0; r < 8; ++r) {
      int m = mbase + r;
      if (m < Mstore) {
        long long idx = coff + (long long)m * ldc + n;
        float v = acc[t][r] + bv;
        if (EP == 1)      ((float*)Cout)[idx] = v + Res[idx];
        else if (EP == 2) ((u16*)Cout)[idx] = f2bf(v);
        else {
          float ge = 0.5f * v * (1.0f + erff(v * 0.70710678118654752f));
          ((u16*)Cout)[idx] = f2bf(ge);
        }
      }
    }
  }
}

// ---------------------------------------------------------------------------
// Fused attention scores + softmax: one wave per (b, head, 16-row q-tile).
// scores = (Q @ K^T) * 0.125, masked past key 256, softmax, probs -> bf16
// probs layout: [B*NH][NPAD_ rows][KPAD_ cols]
// ---------------------------------------------------------------------------
__global__ __launch_bounds__(32) void attn_scores_softmax(const u16* __restrict__ qkv,
                                                          u16* __restrict__ probs) {
  int bid = blockIdx.x;
  int qt = bid % 17;
  int bh = bid / 17;
  int h = bh % NHH_;
  int b = bh / NHH_;
  int lane = threadIdx.x;
  int l16 = lane & 15;
  int lh = lane >> 4;
  int q0 = qt * 16;

  int qr = q0 + l16; if (qr > NN_ - 1) qr = NN_ - 1;
  const u16* qbase = qkv + ((long long)(b * NN_ + qr)) * (3 * DD_) + h * HD_;
  v16bf a0 = frag2(qbase + lh * 8,      qbase + 16 + lh * 8);
  v16bf a1 = frag2(qbase + 32 + lh * 8, qbase + 48 + lh * 8);

  v8f acc[17];
#pragma unroll
  for (int t = 0; t < 17; ++t) acc[t] = v8f{0, 0, 0, 0, 0, 0, 0, 0};

#pragma unroll
  for (int kt = 0; kt < 17; ++kt) {
    int kr = kt * 16 + l16; if (kr > NN_ - 1) kr = NN_ - 1;
    const u16* kb = qkv + ((long long)(b * NN_ + kr)) * (3 * DD_) + DD_ + h * HD_ + lh * 8;
    v16bf b0 = frag2(kb,      kb + 16);
    v16bf b1 = frag2(kb + 32, kb + 48);
    acc[kt] = __builtin_amdgcn_wmma_f32_16x16x32_bf16(false, a0, false, b0,
                                                      (short)0, acc[kt], false, false);
    acc[kt] = __builtin_amdgcn_wmma_f32_16x16x32_bf16(false, a1, false, b1,
                                                      (short)0, acc[kt], false, false);
  }

  const float scale = 0.125f;  // 1/sqrt(64)
  float rmax[8], rsum[8], rinv[8];
#pragma unroll
  for (int r = 0; r < 8; ++r) rmax[r] = -3.0e38f;
#pragma unroll
  for (int kt = 0; kt < 17; ++kt) {
    int key = kt * 16 + l16;
    bool ok = (key <= NN_ - 1);
#pragma unroll
    for (int r = 0; r < 8; ++r) {
      float v = ok ? acc[kt][r] * scale : -3.0e38f;
      acc[kt][r] = v;
      rmax[r] = fmaxf(rmax[r], v);
    }
  }
#pragma unroll
  for (int r = 0; r < 8; ++r) {
#pragma unroll
    for (int m = 8; m; m >>= 1) rmax[r] = fmaxf(rmax[r], __shfl_xor(rmax[r], m, 32));
    rsum[r] = 0.f;
  }
#pragma unroll
  for (int kt = 0; kt < 17; ++kt) {
#pragma unroll
    for (int r = 0; r < 8; ++r) {
      float e = __expf(acc[kt][r] - rmax[r]);
      acc[kt][r] = e;
      rsum[r] += e;
    }
  }
#pragma unroll
  for (int r = 0; r < 8; ++r) {
#pragma unroll
    for (int m = 8; m; m >>= 1) rsum[r] += __shfl_xor(rsum[r], m, 32);
    rinv[r] = 1.f / rsum[r];
  }

  long long base = ((long long)bh * NPAD_ + q0) * KPAD_;
#pragma unroll
  for (int r = 0; r < 8; ++r) {
    long long rowoff = base + (long long)(r + lh * 8) * KPAD_;
#pragma unroll
    for (int kt = 0; kt < 17; ++kt)
      probs[rowoff + kt * 16 + l16] = f2bf(acc[kt][r] * rinv[r]);
    probs[rowoff + NPAD_ + l16] = 0;  // zero-pad cols 272..287
  }
}

// ---------------------------------------------------------------------------
// Final head: out[b,c] = hcls[b,:] . head_w[:,c] + head_b[c]
// ---------------------------------------------------------------------------
__global__ void head_kernel(const float* __restrict__ hcls,
                            const float* __restrict__ w,
                            const float* __restrict__ b,
                            float* __restrict__ out) {
  int idx = blockIdx.x * 256 + threadIdx.x;
  if (idx >= BB_ * NC_) return;
  int bb = idx / NC_, c = idx % NC_;
  float acc = b[c];
  const float* hr = hcls + (long long)bb * DD_;
  for (int d = 0; d < DD_; ++d) acc += hr[d] * w[(long long)d * NC_ + c];
  out[idx] = acc;
}

// ---------------------------------------------------------------------------
// Launch
// ---------------------------------------------------------------------------
extern "C" void kernel_launch(void* const* d_in, const int* in_sizes, int n_in,
                              void* d_out, int out_size, void* d_ws, size_t ws_size,
                              hipStream_t stream) {
  (void)in_sizes; (void)n_in; (void)out_size; (void)ws_size;

  const float* img      = (const float*)d_in[0];
  const int*   segments = (const int*)d_in[1];
  const float* w_fe     = (const float*)d_in[2];
  const float* b_fe     = (const float*)d_in[3];
  const float* cls_tok  = (const float*)d_in[4];
  const float* pos      = (const float*)d_in[5];
  const float* ln1_g    = (const float*)d_in[6];
  const float* ln1_b    = (const float*)d_in[7];
  const float* qkv_w    = (const float*)d_in[8];
  const float* qkv_b    = (const float*)d_in[9];
  const float* proj_w   = (const float*)d_in[10];
  const float* proj_b   = (const float*)d_in[11];
  const float* ln2_g    = (const float*)d_in[12];
  const float* ln2_b    = (const float*)d_in[13];
  const float* mlp_w1   = (const float*)d_in[14];
  const float* mlp_b1   = (const float*)d_in[15];
  const float* mlp_w2   = (const float*)d_in[16];
  const float* mlp_b2   = (const float*)d_in[17];
  const float* norm_g   = (const float*)d_in[18];
  const float* norm_b   = (const float*)d_in[19];
  const float* head_w   = (const float*)d_in[20];
  const float* head_b   = (const float*)d_in[21];

  char* ws = (char*)d_ws;
  size_t off = 0;
  auto take = [&](size_t bytes) -> char* {
    off = (off + 255) & ~(size_t)255;
    char* p = ws + off;
    off += bytes;
    return p;
  };

  u16* wt_qkv  = (u16*)take((size_t)LL_ * 3 * DD_ * DD_ * 2);   // [L][2304][768]
  u16* wt_proj = (u16*)take((size_t)LL_ * DD_ * DD_ * 2);       // [L][768][768]
  u16* wt_w1   = (u16*)take((size_t)LL_ * MLP_ * DD_ * 2);      // [L][3072][768]
  u16* wt_w2   = (u16*)take((size_t)LL_ * DD_ * MLP_ * 2);      // [L][768][3072]
  float* x     = (float*)take((size_t)MM_ * DD_ * 4);
  u16* hbuf    = (u16*)take((size_t)MM_ * DD_ * 2);
  u16* qkvbuf  = (u16*)take((size_t)MM_ * 3 * DD_ * 2);
  u16* probs   = (u16*)take((size_t)BB_ * NHH_ * NPAD_ * KPAD_ * 2);
  u16* vT      = (u16*)take((size_t)BB_ * NHH_ * HD_ * KPAD_ * 2);
  u16* obuf    = (u16*)take((size_t)MM_ * DD_ * 2);
  u16* m1buf   = (u16*)take((size_t)MM_ * MLP_ * 2);
  float* segsum = (float*)take((size_t)BB_ * SS_ * 3 * 4);
  float* hcls   = (float*)take((size_t)BB_ * DD_ * 4);

  // --- superpixel tokenization ---
  zero_f32<<<(BB_ * SS_ * 3 + 255) / 256, 256, 0, stream>>>(segsum, BB_ * SS_ * 3);
  seg_sum_kernel<<<(BB_ * HW_ + 255) / 256, 256, 0, stream>>>(img, segments, segsum);
  tokens_kernel<<<(int)(((long long)MM_ * DD_ + 255) / 256), 256, 0, stream>>>(
      segsum, w_fe, b_fe, cls_tok, pos, x);

  // --- weight conversion (fp32 [K][N] -> bf16 [N][K]) ---
  {
    long long t;
    t = (long long)LL_ * DD_ * 3 * DD_;
    cvt_transpose<<<(int)((t + 255) / 256), 256, 0, stream>>>(qkv_w, wt_qkv, DD_, 3 * DD_, t);
    t = (long long)LL_ * DD_ * DD_;
    cvt_transpose<<<(int)((t + 255) / 256), 256, 0, stream>>>(proj_w, wt_proj, DD_, DD_, t);
    t = (long long)LL_ * DD_ * MLP_;
    cvt_transpose<<<(int)((t + 255) / 256), 256, 0, stream>>>(mlp_w1, wt_w1, DD_, MLP_, t);
    t = (long long)LL_ * MLP_ * DD_;
    cvt_transpose<<<(int)((t + 255) / 256), 256, 0, stream>>>(mlp_w2, wt_w2, MLP_, DD_, t);
  }

  const int MT = (MM_ + 127) / 128;  // 65
  for (int l = 0; l < LL_; ++l) {
    // LN1 -> h (bf16)
    ln_rows<false><<<(MM_ + 7) / 8, 256, 0, stream>>>(x, DD_, MM_, ln1_g + l * DD_,
                                                      ln1_b + l * DD_, hbuf, DD_);
    // QKV = h @ Wqkv + b  (bf16 out)
    gemm_bf16<2><<<dim3(3 * DD_ / 64, MT, 1), 256, 0, stream>>>(
        hbuf, DD_, 0, wt_qkv + (size_t)l * 3 * DD_ * DD_, DD_, 0,
        qkv_b + l * 3 * DD_, qkvbuf, 3 * DD_, nullptr, 0, 0, 1,
        MM_, MM_, 3 * DD_, DD_);
    // scores + softmax -> probs
    attn_scores_softmax<<<BB_ * NHH_ * 17, 32, 0, stream>>>(qkvbuf, probs);
    // V transpose
    build_vt<<<(int)(((long long)BB_ * NHH_ * HD_ * KPAD_ + 255) / 256), 256, 0, stream>>>(
        qkvbuf, vT);
    // O = probs @ V (batched over B*NH, bf16 out scattered into heads)
    gemm_bf16<2><<<dim3(1, (NPAD_ + 127) / 128, BB_ * NHH_), 256, 0, stream>>>(
        probs, KPAD_, (long long)NPAD_ * KPAD_, vT, KPAD_, (long long)HD_ * KPAD_,
        nullptr, obuf, DD_, nullptr, (long long)NN_ * DD_, HD_, NHH_,
        NPAD_, NN_, HD_, KPAD_);
    // x = x + O @ Wproj + b  (f32 in-place)
    gemm_bf16<1><<<dim3(DD_ / 64, MT, 1), 256, 0, stream>>>(
        obuf, DD_, 0, wt_proj + (size_t)l * DD_ * DD_, DD_, 0,
        proj_b + l * DD_, x, DD_, x, 0, 0, 1, MM_, MM_, DD_, DD_);
    // LN2 -> h
    ln_rows<false><<<(MM_ + 7) / 8, 256, 0, stream>>>(x, DD_, MM_, ln2_g + l * DD_,
                                                      ln2_b + l * DD_, hbuf, DD_);
    // m1 = gelu(h @ W1 + b1) (bf16)
    gemm_bf16<3><<<dim3(MLP_ / 64, MT, 1), 256, 0, stream>>>(
        hbuf, DD_, 0, wt_w1 + (size_t)l * MLP_ * DD_, DD_, 0,
        mlp_b1 + l * MLP_, m1buf, MLP_, nullptr, 0, 0, 1, MM_, MM_, MLP_, DD_);
    // x = x + m1 @ W2 + b2
    gemm_bf16<1><<<dim3(DD_ / 64, MT, 1), 256, 0, stream>>>(
        m1buf, MLP_, 0, wt_w2 + (size_t)l * DD_ * MLP_, MLP_, 0,
        mlp_b2 + l * DD_, x, DD_, x, 0, 0, 1, MM_, MM_, DD_, MLP_);
  }

  // final LN on cls rows -> hcls (f32), then head
  ln_rows<true><<<(BB_ + 7) / 8, 256, 0, stream>>>(x, (long long)NN_ * DD_, BB_,
                                                   norm_g, norm_b, hcls, DD_);
  head_kernel<<<(BB_ * NC_ + 255) / 256, 256, 0, stream>>>(hcls, head_w, head_b,
                                                           (float*)d_out);
}
